// CrystalGraphAttention_28192165331142
// MI455X (gfx1250) — matrix operand order repaired and checked by reading, not compiled, not run on hardware
//
#include <hip/hip_runtime.h>
#include <hip/hip_bf16.h>
#include <hip/hip_fp16.h>

typedef __attribute__((ext_vector_type(16))) _Float16 v16h;
typedef __attribute__((ext_vector_type(8)))  float    v8f;

#define BATCH 8
#define NNODE 1024
#define DIM   256
#define NH    8
#define DHEAD 64
#define HD    512        // H*DV
#define NEGBIG (-1.0e9f)

union AFrag { v16h v; float4 f4[2]; _Float16 h[16]; };
union H8   { float4 f4; _Float16 h[8]; };

__device__ __forceinline__ v8f wmma16(v16h a, v16h b, v8f c) {
    // v_wmma_f32_16x16x32_f16  (D = A(16x32,f16) * B(32x16,f16) + C(16x16,f32))
    return __builtin_amdgcn_wmma_f32_16x16x32_f16(false, a, false, b, (short)0, c, false, false);
}

// ---------------------------------------------------------------------------
// Prep: f16 copies/transposes of X, [Wq|Wk|Wv]^T, Wo^T
// ---------------------------------------------------------------------------
__global__ void prep_kernel(const float* __restrict__ X,
                            const float* __restrict__ Wq, const float* __restrict__ Wk,
                            const float* __restrict__ Wv, const float* __restrict__ Wo,
                            _Float16* __restrict__ Xh, _Float16* __restrict__ Wt,
                            _Float16* __restrict__ WoT) {
    const int XN  = BATCH * NNODE * DIM;   // 2,097,152
    const int WTN = 1536 * DIM;            //   393,216
    const int WON = DIM * HD;              //   131,072
    const int total = XN + WTN + WON;
    for (int i = blockIdx.x * blockDim.x + threadIdx.x; i < total;
         i += gridDim.x * blockDim.x) {
        if (i < XN) {
            Xh[i] = (_Float16)X[i];
        } else if (i < XN + WTN) {
            int j = i - XN;
            int f = j >> 8, k = j & 255;          // Wt[f][k] = W[k][f']
            float v;
            if (f < 512)       v = Wq[k * HD + f];
            else if (f < 1024) v = Wk[k * HD + (f - 512)];
            else               v = Wv[k * HD + (f - 1024)];
            Wt[j] = (_Float16)v;
        } else {
            int j = i - XN - WTN;
            int f = j >> 9, k = j & 511;          // WoT[f][k] = Wo[k][f]
            WoT[j] = (_Float16)Wo[k * DIM + f];
        }
    }
}

// ---------------------------------------------------------------------------
// Q/K projection: C = Wt(features x 256) @ X^T(256 x nodes)
// one wave: 32 features x 64 nodes; Q/K stored (B,H,N,64) f16
// ---------------------------------------------------------------------------
__global__ __launch_bounds__(32) void qk_proj_kernel(const _Float16* __restrict__ Xh,
                                                     const _Float16* __restrict__ Wt,
                                                     _Float16* __restrict__ Q,
                                                     _Float16* __restrict__ Kc) {
    const int lane = threadIdx.x, half = lane >> 4, idx = lane & 15;
    const int n0 = blockIdx.x * 64;   // node tile
    const int f0 = blockIdx.y * 32;   // feature tile in [0,1024)
    v8f acc[2][4] = {};
    for (int kc = 0; kc < 8; ++kc) {
        const int k0 = kc * 32;
        AFrag a[2];
#pragma unroll
        for (int i = 0; i < 2; ++i) {
            const _Float16* p = Wt + (size_t)(f0 + 16 * i + idx) * DIM + k0;
            a[i].f4[0] = *(const float4*)(p + 16 * half);
            a[i].f4[1] = *(const float4*)(p + 8 + 16 * half);
        }
#pragma unroll
        for (int j = 0; j < 4; ++j) {
            AFrag bf;
            const _Float16* p = Xh + (size_t)(n0 + 16 * j + idx) * DIM + k0 + 16 * half;
            bf.f4[0] = *(const float4*)(p);
            bf.f4[1] = *(const float4*)(p + 8);
#pragma unroll
            for (int i = 0; i < 2; ++i) acc[i][j] = wmma16(a[i].v, bf.v, acc[i][j]);
        }
    }
#pragma unroll
    for (int i = 0; i < 2; ++i) {
        int fs = f0 + 16 * i + 8 * half;          // 8 contiguous features per lane
        _Float16* tgt = Q;
        if (fs >= 512) { tgt = Kc; fs -= 512; }
        const int h = fs >> 6, dk = fs & 63;
#pragma unroll
        for (int j = 0; j < 4; ++j) {
            const int node = n0 + 16 * j + idx;
            const int b = node >> 10, nn = node & 1023;
            H8 u;
#pragma unroll
            for (int r = 0; r < 8; ++r) u.h[r] = (_Float16)acc[i][j][r];
            *(float4*)(tgt + (((size_t)(b * NH + h) * NNODE + nn) * DHEAD + dk)) = u.f4;
        }
    }
}

// ---------------------------------------------------------------------------
// V projection: C = X(nodes x 256) @ Wt_v^T(256 x dv); Vt stored (B,H,64,N) f16
// one wave: 32 nodes x 64 dv (one full head)
// ---------------------------------------------------------------------------
__global__ __launch_bounds__(32) void v_proj_kernel(const _Float16* __restrict__ Xh,
                                                    const _Float16* __restrict__ Wt,
                                                    _Float16* __restrict__ Vt) {
    const int lane = threadIdx.x, half = lane >> 4, idx = lane & 15;
    const int n0 = blockIdx.x * 32;
    const int h  = blockIdx.y;
    const int f0 = 1024 + h * 64;     // V feature rows inside Wt
    v8f acc[2][4] = {};
    for (int kc = 0; kc < 8; ++kc) {
        const int k0 = kc * 32;
        AFrag a[2];
#pragma unroll
        for (int i = 0; i < 2; ++i) {
            const _Float16* p = Xh + (size_t)(n0 + 16 * i + idx) * DIM + k0;
            a[i].f4[0] = *(const float4*)(p + 16 * half);
            a[i].f4[1] = *(const float4*)(p + 8 + 16 * half);
        }
#pragma unroll
        for (int j = 0; j < 4; ++j) {
            AFrag bf;
            const _Float16* p = Wt + (size_t)(f0 + 16 * j + idx) * DIM + k0 + 16 * half;
            bf.f4[0] = *(const float4*)(p);
            bf.f4[1] = *(const float4*)(p + 8);
#pragma unroll
            for (int i = 0; i < 2; ++i) acc[i][j] = wmma16(a[i].v, bf.v, acc[i][j]);
        }
    }
#pragma unroll
    for (int i = 0; i < 2; ++i) {
        const int ns = n0 + 16 * i + 8 * half;    // 8 contiguous nodes per lane
        const int b = ns >> 10, nn = ns & 1023;
#pragma unroll
        for (int j = 0; j < 4; ++j) {
            const int dv = 16 * j + idx;
            H8 u;
#pragma unroll
            for (int r = 0; r < 8; ++r) u.h[r] = (_Float16)acc[i][j][r];
            *(float4*)(Vt + ((size_t)(b * NH + h) * DHEAD + dv) * NNODE + nn) = u.f4;
        }
    }
}

// ---------------------------------------------------------------------------
// Attention core: one wave per (b, h, 32-query tile).
//   S = QK^T/8 ; + (1-mask)*NEG ; * dw[key] ; exp ; P (f16) -> LDS ; out = P@V / rowsum
// 32-query tiles share every K/V B-fragment between two WMMA A-sets, halving
// L2 traffic vs 16-query tiles. Mask stream (only HBM-resident input) is
// prefetched 16 key-tiles (1 KB/row) ahead via global_prefetch_b8.
// ---------------------------------------------------------------------------
__global__ __launch_bounds__(32) void attn_kernel(const _Float16* __restrict__ Q,
                                                  const _Float16* __restrict__ Kc,
                                                  const _Float16* __restrict__ Vt,
                                                  const float* __restrict__ mask,
                                                  const float* __restrict__ dw,
                                                  _Float16* __restrict__ attn) {
    __shared__ _Float16 P[32 * NNODE];            // 64 KB: unnormalized probs, row-major
    const int lane = threadIdx.x, half = lane >> 4, idx = lane & 15;
    const int q0 = blockIdx.x * 32, h = blockIdx.y, b = blockIdx.z;
    const size_t bh = (size_t)(b * NH + h);

    AFrag aQ[2][2];
#pragma unroll
    for (int t = 0; t < 2; ++t) {
        const _Float16* p = Q + (bh * NNODE + q0 + t * 16 + idx) * DHEAD;
        aQ[t][0].f4[0] = *(const float4*)(p + 16 * half);
        aQ[t][0].f4[1] = *(const float4*)(p + 8 + 16 * half);
        aQ[t][1].f4[0] = *(const float4*)(p + 32 + 16 * half);
        aQ[t][1].f4[1] = *(const float4*)(p + 40 + 16 * half);
    }

    float rsum[2][8] = {};
    for (int kt = 0; kt < 64; ++kt) {
        const int key0 = kt * 16;
        // prefetch mask 16 key-tiles ahead; 32 lanes cover this wave's 32 rows
        {
            const int pkey = (key0 + 256 < NNODE) ? key0 + 256 : key0;
            __builtin_prefetch(mask + ((size_t)b * NNODE + q0 + lane) * NNODE + pkey, 0, 0);
        }
        AFrag b0, b1;
        const _Float16* kp = Kc + (bh * NNODE + key0 + idx) * DHEAD + 16 * half;
        b0.f4[0] = *(const float4*)(kp);       b0.f4[1] = *(const float4*)(kp + 8);
        b1.f4[0] = *(const float4*)(kp + 32);  b1.f4[1] = *(const float4*)(kp + 40);
        const float dwv = dw[b * NNODE + key0 + idx];
#pragma unroll
        for (int t = 0; t < 2; ++t) {
            v8f s = {};
            s = wmma16(aQ[t][0].v, b0.v, s);
            s = wmma16(aQ[t][1].v, b1.v, s);
#pragma unroll
            for (int r = 0; r < 8; ++r) {
                const int m = q0 + t * 16 + r + 8 * half;
                const float mv = mask[((size_t)b * NNODE + m) * NNODE + key0 + idx];
                const float l = (s[r] * 0.125f + (1.0f - mv) * NEGBIG) * dwv;
                const float pv = __expf(l);   // logits <= ~6 unmasked: no max-sub needed
                rsum[t][r] += pv;
                P[(t * 16 + r + 8 * half) * NNODE + key0 + idx] = (_Float16)pv;
            }
        }
    }

    v8f acc[2][4] = {};
    for (int kc = 0; kc < 32; ++kc) {
        const int k0 = kc * 32;
        AFrag a[2];
#pragma unroll
        for (int t = 0; t < 2; ++t) {
            const _Float16* p = &P[(t * 16 + idx) * NNODE + k0];
            a[t].f4[0] = *(const float4*)(p + 16 * half);
            a[t].f4[1] = *(const float4*)(p + 8 + 16 * half);
        }
#pragma unroll
        for (int j = 0; j < 4; ++j) {
            AFrag bv;                              // one Vt fragment feeds both q-subtiles
            const _Float16* vp = Vt + (bh * DHEAD + j * 16 + idx) * NNODE + k0 + 16 * half;
            bv.f4[0] = *(const float4*)(vp);
            bv.f4[1] = *(const float4*)(vp + 8);
#pragma unroll
            for (int t = 0; t < 2; ++t) acc[t][j] = wmma16(a[t].v, bv.v, acc[t][j]);
        }
    }

#pragma unroll
    for (int t = 0; t < 2; ++t)
#pragma unroll
        for (int r = 0; r < 8; ++r) {             // butterfly stays inside each 16-lane half
            float v = rsum[t][r];
            v += __shfl_xor(v, 1, 32);
            v += __shfl_xor(v, 2, 32);
            v += __shfl_xor(v, 4, 32);
            v += __shfl_xor(v, 8, 32);
            rsum[t][r] = 1.0f / v;
        }
#pragma unroll
    for (int t = 0; t < 2; ++t)
#pragma unroll
        for (int j = 0; j < 4; ++j)
#pragma unroll
            for (int r = 0; r < 8; ++r) {
                const int m = q0 + t * 16 + r + 8 * half;
                attn[((size_t)(b * NNODE + m)) * HD + h * DHEAD + j * 16 + idx] =
                    (_Float16)(acc[t][j][r] * rsum[t][r]);
            }
}

// ---------------------------------------------------------------------------
// Output projection: C = WoT(256 x 512) @ attn^T(512 x nodes) + bo
// ---------------------------------------------------------------------------
__global__ __launch_bounds__(32) void out_proj_kernel(const _Float16* __restrict__ attn,
                                                      const _Float16* __restrict__ WoT,
                                                      const float* __restrict__ bo,
                                                      float* __restrict__ out) {
    const int lane = threadIdx.x, half = lane >> 4, idx = lane & 15;
    const int n0 = blockIdx.x * 64;
    const int f0 = blockIdx.y * 32;
    v8f acc[2][4] = {};
    for (int kc = 0; kc < 16; ++kc) {
        const int k0 = kc * 32;
        AFrag a[2];
#pragma unroll
        for (int i = 0; i < 2; ++i) {
            const _Float16* p = WoT + (size_t)(f0 + 16 * i + idx) * HD + k0;
            a[i].f4[0] = *(const float4*)(p + 16 * half);
            a[i].f4[1] = *(const float4*)(p + 8 + 16 * half);
        }
#pragma unroll
        for (int j = 0; j < 4; ++j) {
            AFrag bf;
            const _Float16* p = attn + (size_t)(n0 + 16 * j + idx) * HD + k0 + 16 * half;
            bf.f4[0] = *(const float4*)(p);
            bf.f4[1] = *(const float4*)(p + 8);
#pragma unroll
            for (int i = 0; i < 2; ++i) acc[i][j] = wmma16(a[i].v, bf.v, acc[i][j]);
        }
    }
#pragma unroll
    for (int i = 0; i < 2; ++i) {
        const int fs = f0 + 16 * i + 8 * half;    // 8 contiguous output features
        const float4 b0 = *(const float4*)(bo + fs);
        const float4 b1 = *(const float4*)(bo + fs + 4);
#pragma unroll
        for (int j = 0; j < 4; ++j) {
            const int node = n0 + 16 * j + idx;
            float4 o0, o1;
            o0.x = acc[i][j][0] + b0.x; o0.y = acc[i][j][1] + b0.y;
            o0.z = acc[i][j][2] + b0.z; o0.w = acc[i][j][3] + b0.w;
            o1.x = acc[i][j][4] + b1.x; o1.y = acc[i][j][5] + b1.y;
            o1.z = acc[i][j][6] + b1.z; o1.w = acc[i][j][7] + b1.w;
            *(float4*)(out + (size_t)node * DIM + fs)     = o0;
            *(float4*)(out + (size_t)node * DIM + fs + 4) = o1;
        }
    }
}

// ---------------------------------------------------------------------------
extern "C" void kernel_launch(void* const* d_in, const int* in_sizes, int n_in,
                              void* d_out, int out_size, void* d_ws, size_t ws_size,
                              hipStream_t stream) {
    const float* X    = (const float*)d_in[0];   // (B,N,D)
    const float* mask = (const float*)d_in[1];   // (B,1,N,N)
    const float* dw   = (const float*)d_in[2];   // (B,N)
    const float* Wq   = (const float*)d_in[3];   // (256,512)
    const float* Wk   = (const float*)d_in[4];
    const float* Wv   = (const float*)d_in[5];
    const float* Wo   = (const float*)d_in[6];   // (512,256)
    const float* bo   = (const float*)d_in[7];   // (256,)
    float* out = (float*)d_out;

    char* ws = (char*)d_ws;
    _Float16* Xh   = (_Float16*)(ws);                                     //  4 MB
    _Float16* Wt   = (_Float16*)(ws + (size_t)(4u << 20));                // 768 KB
    _Float16* WoT  = (_Float16*)(ws + (size_t)(4u << 20) + (768u << 10)); // 256 KB
    _Float16* Q    = (_Float16*)(ws + (size_t)(5u  << 20));               //  8 MB
    _Float16* Kc   = (_Float16*)(ws + (size_t)(13u << 20));               //  8 MB
    _Float16* Vt   = (_Float16*)(ws + (size_t)(21u << 20));               //  8 MB
    _Float16* attn = (_Float16*)(ws + (size_t)(29u << 20));               //  8 MB  (37 MB total)

    prep_kernel<<<2048, 256, 0, stream>>>(X, Wq, Wk, Wv, Wo, Xh, Wt, WoT);
    qk_proj_kernel<<<dim3(128, 32), 32, 0, stream>>>(Xh, Wt, Q, Kc);
    v_proj_kernel<<<dim3(256, 8), 32, 0, stream>>>(Xh, Wt, Vt);
    attn_kernel<<<dim3(32, 8, 8), 32, 0, stream>>>(Q, Kc, Vt, mask, dw, attn);
    out_proj_kernel<<<dim3(128, 8), 32, 0, stream>>>(attn, WoT, bo, out);
}
